// LiteMLA_25065429139566
// MI455X (gfx1250) — matrix operand order repaired and checked by reading, not compiled, not run on hardware
//
#include <hip/hip_runtime.h>
#include <hip/hip_bf16.h>

typedef __attribute__((ext_vector_type(16))) __bf16 v16bf;
typedef __attribute__((ext_vector_type(8)))  __bf16 v8bf;
typedef __attribute__((ext_vector_type(4)))  __bf16 v4bf;
typedef __attribute__((ext_vector_type(8)))  float  v8f;

#define NTOK   4096
#define BN_EPS 1e-5f

__device__ __forceinline__ __bf16 tobf(float f) { return (__bf16)f; }

// ---- optional CDNA5 async global->LDS path (guarded; sync fallback) -------
#if defined(__has_builtin)
#if __has_builtin(__builtin_amdgcn_global_load_async_to_lds_b128)
#define HAVE_ASYNC_LDS 1
#endif
#endif

#ifdef HAVE_ASYNC_LDS
typedef int v4i __attribute__((vector_size(16)));
typedef __attribute__((address_space(1))) v4i* g_v4i;
typedef __attribute__((address_space(3))) v4i* l_v4i;
__device__ __forceinline__ void async_cp16(const void* g, void* l) {
    void* gp = const_cast<void*>(g);
    __builtin_amdgcn_global_load_async_to_lds_b128((g_v4i)gp, (l_v4i)l, 0, 0);
}
#if __has_builtin(__builtin_amdgcn_s_wait_asynccnt)
#define WAIT_ASYNC() __builtin_amdgcn_s_wait_asynccnt(0)
#else
#define WAIT_ASYNC() asm volatile("s_wait_asynccnt 0" ::: "memory")
#endif
#else
__device__ __forceinline__ void async_cp16(const void* g, void* l) {
    *(float4*)l = *(const float4*)g;
}
#define WAIT_ASYNC()
#endif

// ---------------------------------------------------------------- BN params
__global__ void bn_prep(const float* __restrict__ qg, const float* __restrict__ qb,
                        const float* __restrict__ qm, const float* __restrict__ qv,
                        const float* __restrict__ pg, const float* __restrict__ pb,
                        const float* __restrict__ pm, const float* __restrict__ pv,
                        float* __restrict__ sc) {
    int t = blockIdx.x * blockDim.x + threadIdx.x;
    if (t < 768) {
        float s = qg[t] * rsqrtf(qv[t] + BN_EPS);
        sc[t]       = s;
        sc[768 + t] = qb[t] - qm[t] * s;
    } else if (t < 1024) {
        int c = t - 768;
        float s = pg[c] * rsqrtf(pv[c] + BN_EPS);
        sc[1536 + c]       = s;
        sc[1536 + 256 + c] = pb[c] - pm[c] * s;
    }
}

// -------------------------------------------------- tiled WMMA GEMM + BN
// out[b,o,n] = BN( sum_c w[o,c] * in[b,c,n] )
// Tile 64(M) x 64(N) x 32(K); 256 threads = 8 waves, 2 subtiles each.
// LDS tiles are bf16; X tile stored transposed [n][k] so B-fragments are
// two contiguous 8-element runs, as are A-fragments.
__global__ void gemm_bn(const float* __restrict__ in, int inBstride,
                        const float* __restrict__ w, int K, int otiles,
                        const float* __restrict__ scale, const float* __restrict__ shift,
                        float* __restrict__ out, int outBstride) {
    __shared__ __bf16 Wt[64][40];   // [o][k], row stride 80B (16B aligned)
    __shared__ __bf16 Xt[64][40];   // [n][k] transposed

    const int bpb   = otiles * (NTOK / 64);
    const int b     = blockIdx.x / bpb;
    const int rem   = blockIdx.x % bpb;
    const int obase = (rem / (NTOK / 64)) * 64;
    const int nbase = (rem % (NTOK / 64)) * 64;

    const int lane = threadIdx.x & 31;
    const int wave = threadIdx.x >> 5;
    const int m    = lane & 15;
    const int hi   = lane >> 4;

    v8f acc0 = {0.f,0.f,0.f,0.f,0.f,0.f,0.f,0.f};
    v8f acc1 = acc0;

    for (int kb = 0; kb < K; kb += 32) {
        // prefetch next K-chunk (global_prefetch_b8)
        if (kb + 32 < K) {
            __builtin_prefetch(w + (size_t)(obase + (threadIdx.x >> 3)) * K + kb + 32, 0, 1);
            __builtin_prefetch(in + ((size_t)b * inBstride + kb + 32 + (threadIdx.x >> 4)) * NTOK + nbase, 0, 1);
        }
        // stage W [64 x 32] -> bf16 LDS (layout preserved)
        #pragma unroll
        for (int rr = 0; rr < 2; ++rr) {
            int r = (threadIdx.x >> 3) + rr * 32;
            int c = (threadIdx.x & 7) * 4;
            const float4 wv = *(const float4*)(w + (size_t)(obase + r) * K + kb + c);
            v4bf pk = { tobf(wv.x), tobf(wv.y), tobf(wv.z), tobf(wv.w) };
            *(v4bf*)&Wt[r][c] = pk;
        }
        // stage X [32 x 64] -> bf16 LDS transposed [n][k]
        #pragma unroll
        for (int rr = 0; rr < 2; ++rr) {
            int k = (threadIdx.x >> 4) + rr * 16;
            int n = (threadIdx.x & 15) * 4;
            const float4 xv = *(const float4*)(in + ((size_t)b * inBstride + kb + k) * NTOK + nbase + n);
            Xt[n    ][k] = tobf(xv.x);
            Xt[n + 1][k] = tobf(xv.y);
            Xt[n + 2][k] = tobf(xv.z);
            Xt[n + 3][k] = tobf(xv.w);
        }
        __syncthreads();

        #pragma unroll
        for (int i = 0; i < 2; ++i) {
            int s  = wave * 2 + i;
            int ot = s >> 2, nt = s & 3;
            // A frag: lane holds M=m, K = {grp*16 + hi*8 + 0..7} contiguous
            const v8bf alo = *(const v8bf*)&Wt[ot * 16 + m][hi * 8];
            const v8bf ahi = *(const v8bf*)&Wt[ot * 16 + m][16 + hi * 8];
            v16bf a = __builtin_shufflevector(alo, ahi,
                        0,1,2,3,4,5,6,7,8,9,10,11,12,13,14,15);
            // B frag: lane holds N=m, K = {grp*16 + hi*8 + 0..7} (transposed tile)
            const v8bf blo = *(const v8bf*)&Xt[nt * 16 + m][hi * 8];
            const v8bf bhi = *(const v8bf*)&Xt[nt * 16 + m][16 + hi * 8];
            v16bf bb = __builtin_shufflevector(blo, bhi,
                        0,1,2,3,4,5,6,7,8,9,10,11,12,13,14,15);
            if (i == 0)
                acc0 = __builtin_amdgcn_wmma_f32_16x16x32_bf16(false, a, false, bb, (short)0, acc0, false, false);
            else
                acc1 = __builtin_amdgcn_wmma_f32_16x16x32_bf16(false, a, false, bb, (short)0, acc1, false, false);
        }
        __syncthreads();
    }

    #pragma unroll
    for (int i = 0; i < 2; ++i) {
        int s  = wave * 2 + i;
        int ot = s >> 2, nt = s & 3;
        int n  = nbase + nt * 16 + m;
        v8f acc = (i == 0) ? acc0 : acc1;
        #pragma unroll
        for (int r = 0; r < 8; ++r) {
            int o = obase + ot * 16 + hi * 8 + r;
            out[((size_t)b * outBstride + o) * NTOK + n] = acc[r] * scale[o] + shift[o];
        }
    }
}

// --------------------------------------------- q softmax over tokens (row=4096)
__global__ void softmax_rows(float* __restrict__ ws) {
    __shared__ float red[256];
    const int blk = blockIdx.x;            // 16*256 rows
    const int b = blk >> 8, c = blk & 255;
    float* p = ws + ((size_t)b * 768 + c) * NTOK;
    const int t = threadIdx.x;

    float v[16];
    float mx = -3.4e38f;
    #pragma unroll
    for (int i = 0; i < 16; ++i) { v[i] = p[t + i * 256]; mx = fmaxf(mx, v[i]); }
    red[t] = mx; __syncthreads();
    for (int s = 128; s > 0; s >>= 1) { if (t < s) red[t] = fmaxf(red[t], red[t + s]); __syncthreads(); }
    mx = red[0]; __syncthreads();

    float sum = 0.f;
    #pragma unroll
    for (int i = 0; i < 16; ++i) { v[i] = __expf(v[i] - mx); sum += v[i]; }
    red[t] = sum; __syncthreads();
    for (int s = 128; s > 0; s >>= 1) { if (t < s) red[t] += red[t + s]; __syncthreads(); }
    float inv = __frcp_rn(red[0]);
    #pragma unroll
    for (int i = 0; i < 16; ++i) p[t + i * 256] = v[i] * inv;
}

// --------------------------------------------- k softmax over head_dim (=32)
__global__ void softmax_k(float* __restrict__ ws) {
    const int idx = blockIdx.x * blockDim.x + threadIdx.x;  // 16*8*4096
    const int n  = idx & (NTOK - 1);
    const int bh = idx >> 12;
    const int h = bh & 7, b = bh >> 3;
    float* p = ws + ((size_t)b * 768 + 256 + h * 32) * NTOK + n;

    float v[32], mx = -3.4e38f;
    #pragma unroll
    for (int d = 0; d < 32; ++d) { v[d] = p[(size_t)d * NTOK]; mx = fmaxf(mx, v[d]); }
    float sum = 0.f;
    #pragma unroll
    for (int d = 0; d < 32; ++d) { v[d] = __expf(v[d] - mx); sum += v[d]; }
    float inv = __frcp_rn(sum);
    #pragma unroll
    for (int d = 0; d < 32; ++d) p[(size_t)d * NTOK] = v[d] * inv;
}

// ---------------- context[bh,d,e] = sum_n ksm[d,n]*v[e,n]  (K=4096)
// Tiles of ksm/v (32x32 f32 each) staged to LDS with async global->LDS copies
// (ASYNCcnt path) when available; fragments built from LDS with pk-cvt.
__global__ void ctx_kernel(const float* __restrict__ ws, float* __restrict__ ctx) {
    __shared__ float Kt[32][36];  // row stride 144B (16B aligned)
    __shared__ float Vt[32][36];

    const int bh = blockIdx.x;                 // 128
    const int b = bh >> 3, h = bh & 7;
    const float* kp = ws + ((size_t)b * 768 + 256 + (size_t)h * 32) * NTOK;
    const float* vp = ws + ((size_t)b * 768 + 512 + (size_t)h * 32) * NTOK;

    const int t    = threadIdx.x;              // 128 threads = 4 waves
    const int lane = t & 31, wave = t >> 5;
    const int m = lane & 15, hi = lane >> 4;
    const int dt = wave >> 1, et = wave & 1;

    v8f acc = {0.f,0.f,0.f,0.f,0.f,0.f,0.f,0.f};
    for (int n0 = 0; n0 < NTOK; n0 += 32) {
        // stage 32x32 f32 tiles: 256 x 16B chunks each, 2 passes of 128 threads
        #pragma unroll
        for (int p = 0; p < 2; ++p) {
            int c   = p * 128 + t;
            int row = c >> 3;
            int q4  = (c & 7) * 4;
            async_cp16(kp + (size_t)row * NTOK + n0 + q4, &Kt[row][q4]);
            async_cp16(vp + (size_t)row * NTOK + n0 + q4, &Vt[row][q4]);
        }
        WAIT_ASYNC();
        __syncthreads();

        v16bf a, bb;
        #pragma unroll
        for (int j = 0; j < 8; ++j) {
            int grp = j >> 2, w2 = j & 3;
            int ka  = grp * 16 + hi * 8 + w2 * 2;
            a[2*j]   = tobf(Kt[dt * 16 + m][ka]);
            a[2*j+1] = tobf(Kt[dt * 16 + m][ka + 1]);
            int kb2 = grp * 16 + hi * 8 + w2 * 2;
            bb[2*j]   = tobf(Vt[et * 16 + m][kb2]);
            bb[2*j+1] = tobf(Vt[et * 16 + m][kb2 + 1]);
        }
        acc = __builtin_amdgcn_wmma_f32_16x16x32_bf16(false, a, false, bb, (short)0, acc, false, false);
        __syncthreads();
    }
    #pragma unroll
    for (int r = 0; r < 8; ++r)
        ctx[(size_t)bh * 1024 + (size_t)(dt * 16 + hi * 8 + r) * 32 + et * 16 + m] = acc[r];
}

// ------------------- out[e,n] = sum_d ctx[d,e]*qsm[d,n]; K=32 -> one WMMA/tile
// Result written into the (now dead) k region of ws: channels 256..511.
__global__ void attn_apply(float* __restrict__ ws, const float* __restrict__ ctx) {
    const int lane = threadIdx.x & 31, wave = threadIdx.x >> 5;
    const int tile = blockIdx.x * 8 + wave;          // 65536 tiles
    const int nt = tile & 255;
    const int et = (tile >> 8) & 1;
    const int bh = tile >> 9;
    const int b = bh >> 3, h = bh & 7;
    const int m = lane & 15, hi = lane >> 4;

    const float* qp = ws + ((size_t)b * 768 + (size_t)h * 32) * NTOK;
    const float* cp = ctx + (size_t)bh * 1024;

    v16bf a, bb;
    #pragma unroll
    for (int j = 0; j < 8; ++j) {
        int grp = j >> 2, w2 = j & 3;
        int ka  = grp * 16 + hi * 8 + w2 * 2;        // d for A
        int e   = et * 16 + m;
        a[2*j]   = tobf(cp[(size_t)ka * 32 + e]);
        a[2*j+1] = tobf(cp[(size_t)(ka + 1) * 32 + e]);
        int kb2 = grp * 16 + hi * 8 + w2 * 2;        // d for B
        int n   = nt * 16 + m;
        bb[2*j]   = tobf(qp[(size_t)kb2 * NTOK + n]);
        bb[2*j+1] = tobf(qp[(size_t)(kb2 + 1) * NTOK + n]);
    }
    v8f acc = {0.f,0.f,0.f,0.f,0.f,0.f,0.f,0.f};
    acc = __builtin_amdgcn_wmma_f32_16x16x32_bf16(false, a, false, bb, (short)0, acc, false, false);

    float* op = ws + ((size_t)b * 768 + 256 + (size_t)h * 32) * NTOK;
    #pragma unroll
    for (int r = 0; r < 8; ++r) {
        int e = et * 16 + hi * 8 + r;
        op[(size_t)e * NTOK + nt * 16 + m] = acc[r];
    }
}

// ---------------------------------------------------------------- launcher
extern "C" void kernel_launch(void* const* d_in, const int* in_sizes, int n_in,
                              void* d_out, int out_size, void* d_ws, size_t ws_size,
                              hipStream_t stream) {
    const float* x      = (const float*)d_in[0];
    const float* qkv_w  = (const float*)d_in[1];
    const float* qg     = (const float*)d_in[2];
    const float* qb     = (const float*)d_in[3];
    const float* qm     = (const float*)d_in[4];
    const float* qv     = (const float*)d_in[5];
    const float* proj_w = (const float*)d_in[6];
    const float* pg     = (const float*)d_in[7];
    const float* pb     = (const float*)d_in[8];
    const float* pm     = (const float*)d_in[9];
    const float* pv     = (const float*)d_in[10];
    float* out = (float*)d_out;

    // ws layout (floats): qkv[16*768*4096] | ctx[128*1024] | scales[2048]
    float* ws   = (float*)d_ws;
    float* ctx  = ws + (size_t)16 * 768 * NTOK;
    float* sc   = ctx + (size_t)128 * 1024;

    bn_prep<<<4, 256, 0, stream>>>(qg, qb, qm, qv, pg, pb, pm, pv, sc);

    // qkv = BN(x @ qkv_w^T): M=768, K=256
    gemm_bn<<<16 * 12 * (NTOK / 64), 256, 0, stream>>>(
        x, 256, qkv_w, 256, 12, sc, sc + 768, ws, 768);

    softmax_rows<<<16 * 256, 256, 0, stream>>>(ws);          // q over tokens
    softmax_k<<<16 * 8 * NTOK / 256, 256, 0, stream>>>(ws);  // k over head_dim

    ctx_kernel<<<128, 128, 0, stream>>>(ws, ctx);
    attn_apply<<<8192, 256, 0, stream>>>(ws, ctx);

    // proj = BN(attn_out @ proj_w^T): M=256, K=256; input at ws channel 256..511
    gemm_bn<<<16 * 4 * (NTOK / 64), 256, 0, stream>>>(
        ws + (size_t)256 * NTOK, 768, proj_w, 256, 4, sc + 1536, sc + 1792, out, 256);
}
